// NonlocalBlock_25692494365229
// MI455X (gfx1250) — matrix operand order repaired and checked by reading, not compiled, not run on hardware
//
#include <hip/hip_runtime.h>

#define NB   4
#define CC   256
#define CSs  128
#define NN   4096      // 64*64 tokens
#define EPSBN 1e-5f

typedef __attribute__((ext_vector_type(16))) _Float16 v16h;
typedef __attribute__((ext_vector_type(8)))  _Float16 v8h;
typedef __attribute__((ext_vector_type(8)))  float    v8f;

__device__ __forceinline__ v16h cat8(v8h lo, v8h hi) {
    v16h r;
#pragma unroll
    for (int i = 0; i < 8; ++i) { r[i] = lo[i]; r[i + 8] = hi[i]; }
    return r;
}

__device__ __forceinline__ v8f wmma_f16(v16h a, v16h b, v8f c) {
    // D = A(16x32 f16) * B(32x16 f16) + C(16x16 f32)
    return __builtin_amdgcn_wmma_f32_16x16x32_f16(false, a, false, b,
                                                  (short)0, c, false, false);
}

// ---------------------------------------------------------------------------
// Kernel 0: zero BN accumulators (workspace is poisoned with 0xAA)
// ---------------------------------------------------------------------------
__global__ void zero_stats_kernel(float* p, int n) {
    int i = blockIdx.x * blockDim.x + threadIdx.x;
    if (i < n) p[i] = 0.f;
}

// ---------------------------------------------------------------------------
// Kernel 1: fused Q/K/V 1x1-conv projections.
// Per wave: one 16(token) x 16(cs) output tile, K = C = 256 (8 chunks of 32).
// A = X^T tile (f32 -> f16 on load), B = W^T tile.
// Q,K stored token-major f16 [B][N][CS]; V stored channel-major f16 [B][CS][N].
// ---------------------------------------------------------------------------
__global__ void qkv_proj_kernel(const float* __restrict__ X,
                                const float* __restrict__ Wq, const float* __restrict__ bq,
                                const float* __restrict__ Wk, const float* __restrict__ bk,
                                const float* __restrict__ Wv, const float* __restrict__ bv,
                                _Float16* __restrict__ qh, _Float16* __restrict__ kh,
                                _Float16* __restrict__ vh) {
    const int lane = threadIdx.x & 31;
    const int wave = threadIdx.x >> 5;
    const int g = lane >> 4, ln = lane & 15;

    int t = blockIdx.x * 8 + wave;
    const int cs_t  = t & 7;   t >>= 3;    // CS/16 = 8 tiles
    const int tok_t = t & 255; t >>= 8;    // N/16  = 256 tiles
    const int proj  = t % 3;
    const int b     = t / 3;
    const int n0 = tok_t * 16, cs0 = cs_t * 16;

    const float* Wm = (proj == 0) ? Wq : (proj == 1) ? Wk : Wv;
    const float* bm = (proj == 0) ? bq : (proj == 1) ? bk : bv;
    const float* Xb = X + (size_t)b * CC * NN;

    v8f acc = {};
#pragma unroll
    for (int kc = 0; kc < CC / 32; ++kc) {
        v16h af, bf;
#pragma unroll
        for (int i = 0; i < 8; ++i) {   // A: row M=ln token, K per ISA A-layout
            const int k0 = kc * 32 + 8 * g + i;        // halves 0..7
            const int k1 = kc * 32 + 16 + 8 * g + i;   // halves 8..15
            af[i]     = (_Float16)Xb[(size_t)k0 * NN + n0 + ln];
            af[i + 8] = (_Float16)Xb[(size_t)k1 * NN + n0 + ln];
        }
#pragma unroll
        for (int h = 0; h < 16; ++h)    // B: col N=ln cs, K = 8*g + (h&7) + 16*(h>>3)
            bf[h] = (_Float16)Wm[(size_t)(cs0 + ln) * CC + kc * 32 + 8 * g + (h & 7) + ((h >> 3) << 4)];
        acc = wmma_f16(af, bf, acc);
    }
    const float bias = bm[cs0 + ln];
    if (proj == 2) {
#pragma unroll
        for (int r = 0; r < 8; ++r) {
            const int tok = n0 + r + 8 * g;
            vh[((size_t)b * CSs + cs0 + ln) * NN + tok] = (_Float16)(acc[r] + bias);
        }
    } else {
        _Float16* dst = (proj == 0) ? qh : kh;
#pragma unroll
        for (int r = 0; r < 8; ++r) {
            const int tok = n0 + r + 8 * g;
            dst[((size_t)b * NN + tok) * CSs + cs0 + ln] = (_Float16)(acc[r] + bias);
        }
    }
}

// ---------------------------------------------------------------------------
// Kernel 2: streaming (flash) attention. Block = 8 waves = 128 query rows.
// Per wave: 16 query rows, stream 32 keys/iter:
//   S = Q*K^T (8 wmma), online max via shuffle reduction, P (f16) through a
//   per-wave LDS slab into A-layout, O += P*V (8 wmma), and the softmax
//   denominator accumulated with one extra wmma against an all-ones B matrix
//   (every column of that D-tile is the exact row sum of P).
// ---------------------------------------------------------------------------
__global__ void __launch_bounds__(256)
attention_kernel(const _Float16* __restrict__ qh,
                 const _Float16* __restrict__ kh,
                 const _Float16* __restrict__ vh,
                 _Float16* __restrict__ oh) {
    __shared__ __align__(64) _Float16 Plds[8][16][32];   // 8 KiB, private per wave

    const int lane = threadIdx.x & 31;
    const int wave = threadIdx.x >> 5;
    const int g = lane >> 4, ln = lane & 15;
    const int b  = blockIdx.x >> 5;                 // 32 blocks per batch
    const int q0 = (blockIdx.x & 31) * 128 + wave * 16;

    // Q fragments: A-matrix, 4 chunks over CS=128; two 16B loads per chunk
    v16h qf[4];
    const _Float16* qrow = qh + ((size_t)b * NN + q0 + ln) * CSs;
#pragma unroll
    for (int c = 0; c < 4; ++c) {
        v8h lo = *(const v8h*)(qrow + c * 32 + 16 * g);
        v8h hi = *(const v8h*)(qrow + c * 32 + 16 * g + 8);
        qf[c] = cat8(lo, hi);
    }

    v16h ones;
#pragma unroll
    for (int h = 0; h < 16; ++h) ones[h] = (_Float16)1.0f;

    v8f zero = {};
    v8f o[8];
#pragma unroll
    for (int t = 0; t < 8; ++t) o[t] = zero;
    v8f lacc = zero;                 // per-row softmax denominators (ones-GEMM)
    float mrow[8];
#pragma unroll
    for (int r = 0; r < 8; ++r) mrow[r] = -3.0e38f;

    const _Float16* kbase = kh + (size_t)b * NN * CSs;
    const _Float16* vbase = vh + (size_t)b * CSs * NN;

    for (int j = 0; j < NN; j += 32) {
        // prefetch next iteration's K rows and V columns (global_prefetch_b8)
        if (j + 32 < NN) {
            __builtin_prefetch(kbase + (size_t)(j + 32 + ln) * CSs, 0, 3);
            __builtin_prefetch(kbase + (size_t)(j + 48 + ln) * CSs, 0, 3);
            __builtin_prefetch(vbase + (size_t)(2 * lane) * NN + j + 32, 0, 3);
        }
        v8f s0 = zero, s1 = zero;
#pragma unroll
        for (int c = 0; c < 4; ++c) {
            v16h k0 = *(const v16h*)(kbase + (size_t)(j + ln) * CSs + c * 32 + 16 * g);
            s0 = wmma_f16(qf[c], k0, s0);
        }
#pragma unroll
        for (int c = 0; c < 4; ++c) {
            v16h k1 = *(const v16h*)(kbase + (size_t)(j + 16 + ln) * CSs + c * 32 + 16 * g);
            s1 = wmma_f16(qf[c], k1, s1);
        }
        // online softmax per query row (row r+8g lives in VGPR r across one 16-lane half)
#pragma unroll
        for (int r = 0; r < 8; ++r) {
            float v0 = s0[r], v1 = s1[r];
            float mx = fmaxf(v0, v1);
            mx = fmaxf(mx, __shfl_xor(mx, 1, 32));
            mx = fmaxf(mx, __shfl_xor(mx, 2, 32));
            mx = fmaxf(mx, __shfl_xor(mx, 4, 32));
            mx = fmaxf(mx, __shfl_xor(mx, 8, 32));
            const float mnew = fmaxf(mrow[r], mx);
            const float corr = __expf(mrow[r] - mnew);
            mrow[r] = mnew;
            const float p0 = __expf(v0 - mnew);
            const float p1 = __expf(v1 - mnew);
            lacc[r] *= corr;
#pragma unroll
            for (int t = 0; t < 8; ++t) o[t][r] *= corr;
            Plds[wave][r + 8 * g][ln]      = (_Float16)p0;
            Plds[wave][r + 8 * g][16 + ln] = (_Float16)p1;
        }
        // re-read P in A-fragment layout (same-wave LDS round trip)
        v8h plo = *(const v8h*)&Plds[wave][ln][16 * g];
        v8h phi = *(const v8h*)&Plds[wave][ln][16 * g + 8];
        v16h pf = cat8(plo, phi);
        lacc = wmma_f16(pf, ones, lacc);   // row sums of P -> denominators
#pragma unroll
        for (int t = 0; t < 8; ++t) {   // V channel-major -> contiguous B-frags
            v16h vf = *(const v16h*)(vbase + (size_t)(t * 16 + ln) * NN + j + 16 * g);
            o[t] = wmma_f16(pf, vf, o[t]);
        }
    }
    // normalize and store token-major f16 [B][N][CS]
#pragma unroll
    for (int r = 0; r < 8; ++r) {
        const float inv = 1.0f / lacc[r];
        const size_t rowoff = ((size_t)b * NN + q0 + r + 8 * g) * CSs;
#pragma unroll
        for (int t = 0; t < 8; ++t)
            oh[rowoff + t * 16 + ln] = (_Float16)(o[t][r] * inv);
    }
}

// ---------------------------------------------------------------------------
// Kernel 3: output 1x1 conv (y = Wo*O + bo, f32 out) + per-channel sum/sumsq.
// Per wave: 16 channels x 256 tokens strip (16 WMMA tiles), then shuffle-
// reduce channel partials and one atomicAdd per channel.
// ---------------------------------------------------------------------------
__global__ void outconv_kernel(const _Float16* __restrict__ oh,
                               const float* __restrict__ Wo,
                               const float* __restrict__ bo,
                               float* __restrict__ y,
                               float* __restrict__ gsum,
                               float* __restrict__ gsq) {
    const int lane = threadIdx.x & 31;
    const int wave = threadIdx.x >> 5;
    const int g = lane >> 4, ln = lane & 15;

    int t = blockIdx.x * 8 + wave;
    const int ns     = t & 15; t >>= 4;   // 16 token strips of 256
    const int cstrip = t & 15; t >>= 4;   // C/16 = 16
    const int b = t;
    const int c0 = cstrip * 16, nbase = ns * 256;

    v16h af[4];                            // A = Wo rows, K = CS (4 chunks)
#pragma unroll
    for (int c = 0; c < 4; ++c) {
#pragma unroll
        for (int i = 0; i < 8; ++i) {
            af[c][i]     = (_Float16)Wo[(size_t)(c0 + ln) * CSs + c * 32 + 16 * g + i];
            af[c][i + 8] = (_Float16)Wo[(size_t)(c0 + ln) * CSs + c * 32 + 16 * g + 8 + i];
        }
    }
    float bias[8], sum[8], sq[8];
#pragma unroll
    for (int r = 0; r < 8; ++r) { bias[r] = bo[c0 + r + 8 * g]; sum[r] = 0.f; sq[r] = 0.f; }

    for (int tt = 0; tt < 16; ++tt) {
        const int n0 = nbase + tt * 16;
        v8f acc = {};
#pragma unroll
        for (int c = 0; c < 4; ++c) {
            v16h bf = *(const v16h*)(oh + ((size_t)b * NN + n0 + ln) * CSs + c * 32 + 16 * g);
            acc = wmma_f16(af[c], bf, acc);
        }
#pragma unroll
        for (int r = 0; r < 8; ++r) {
            const float v = acc[r] + bias[r];
            y[((size_t)b * CC + c0 + r + 8 * g) * NN + n0 + ln] = v;
            sum[r] += v;
            sq[r]  += v * v;
        }
    }
#pragma unroll
    for (int r = 0; r < 8; ++r) {
        float s = sum[r], q = sq[r];
        s += __shfl_xor(s, 1, 32); s += __shfl_xor(s, 2, 32);
        s += __shfl_xor(s, 4, 32); s += __shfl_xor(s, 8, 32);
        q += __shfl_xor(q, 1, 32); q += __shfl_xor(q, 2, 32);
        q += __shfl_xor(q, 4, 32); q += __shfl_xor(q, 8, 32);
        if (ln == 0) {
            atomicAdd(&gsum[c0 + r + 8 * g], s);
            atomicAdd(&gsq [c0 + r + 8 * g], q);
        }
    }
}

// ---------------------------------------------------------------------------
// Kernel 4: BN finalize -> per-channel scale/shift
// ---------------------------------------------------------------------------
__global__ void bn_finalize_kernel(const float* __restrict__ gsum,
                                   const float* __restrict__ gsq,
                                   const float* __restrict__ gamma,
                                   const float* __restrict__ beta,
                                   float* __restrict__ scale,
                                   float* __restrict__ shift) {
    const int c = threadIdx.x;
    const float invn = 1.0f / (float)(NB * NN);
    const float mean = gsum[c] * invn;
    const float var  = gsq[c] * invn - mean * mean;
    const float a = gamma[c] * rsqrtf(var + EPSBN);
    scale[c] = a;
    shift[c] = beta[c] - mean * a;
}

// ---------------------------------------------------------------------------
// Kernel 5: out = relu(y*scale + shift) + input   (float4 vectorized)
// ---------------------------------------------------------------------------
__global__ void epilogue_kernel(const float* __restrict__ y,
                                const float* __restrict__ X,
                                const float* __restrict__ scale,
                                const float* __restrict__ shift,
                                float* __restrict__ out) {
    const size_t i = ((size_t)blockIdx.x * blockDim.x + threadIdx.x) * 4;
    const int c = (int)((i / NN) % CC);
    const float a = scale[c], sh = shift[c];
    const float4 yv = *(const float4*)(y + i);
    const float4 xv = *(const float4*)(X + i);
    float4 ov;
    ov.x = fmaxf(yv.x * a + sh, 0.f) + xv.x;
    ov.y = fmaxf(yv.y * a + sh, 0.f) + xv.y;
    ov.z = fmaxf(yv.z * a + sh, 0.f) + xv.z;
    ov.w = fmaxf(yv.w * a + sh, 0.f) + xv.w;
    *(float4*)(out + i) = ov;
}

// ---------------------------------------------------------------------------
extern "C" void kernel_launch(void* const* d_in, const int* in_sizes, int n_in,
                              void* d_out, int out_size, void* d_ws, size_t ws_size,
                              hipStream_t stream) {
    (void)in_sizes; (void)n_in; (void)out_size; (void)ws_size;
    const float* X     = (const float*)d_in[0];
    const float* Wq    = (const float*)d_in[1];
    const float* bq    = (const float*)d_in[2];
    const float* Wk    = (const float*)d_in[3];
    const float* bk    = (const float*)d_in[4];
    const float* Wv    = (const float*)d_in[5];
    const float* bv    = (const float*)d_in[6];
    const float* Wo    = (const float*)d_in[7];
    const float* bo    = (const float*)d_in[8];
    const float* gamma = (const float*)d_in[9];
    const float* beta  = (const float*)d_in[10];
    float* out = (float*)d_out;

    char* ws = (char*)d_ws;
    const size_t HB = (size_t)NB * NN * CSs * sizeof(_Float16);   // 4 MiB each
    _Float16* qh = (_Float16*)(ws);
    _Float16* kh = (_Float16*)(ws + HB);
    _Float16* vh = (_Float16*)(ws + 2 * HB);
    _Float16* oh = (_Float16*)(ws + 3 * HB);
    float* y     = (float*)(ws + 4 * HB);                          // 16 MiB f32
    float* gsum  = (float*)(ws + 4 * HB + (size_t)NB * CC * NN * sizeof(float));
    float* gsq   = gsum + CC;
    float* scale = gsum + 2 * CC;
    float* shift = gsum + 3 * CC;

    zero_stats_kernel<<<2, 256, 0, stream>>>(gsum, 2 * CC);
    qkv_proj_kernel<<<(NB * 3 * (NN / 16) * (CSs / 16)) / 8, 256, 0, stream>>>(
        X, Wq, bq, Wk, bk, Wv, bv, qh, kh, vh);
    attention_kernel<<<NB * (NN / 128), 256, 0, stream>>>(qh, kh, vh, oh);
    outconv_kernel<<<(NB * (CC / 16) * 16) / 8, 256, 0, stream>>>(oh, Wo, bo, y, gsum, gsq);
    bn_finalize_kernel<<<1, CC, 0, stream>>>(gsum, gsq, gamma, beta, scale, shift);
    epilogue_kernel<<<(NB * CC * NN / 4) / 256, 256, 0, stream>>>(y, X, scale, shift, out);
}
